// VisionMamba_15547781611739
// MI455X (gfx1250) — compile-verified, matrix operand order checked by reading, not run on hardware
//
#include <hip/hip_runtime.h>
#include <hip/hip_bf16.h>
#include <cstdint>

#define B_     16
#define L_     197
#define D_     384
#define DEPTH_ 24
#define DI_    768
#define DS_    16
#define DTR_   24
#define NTOK   (B_ * L_)       // 3152 tokens
#define NPATCH 196
#define NCLS   1000

typedef __attribute__((ext_vector_type(16))) _Float16 v16h;
typedef __attribute__((ext_vector_type(8)))  _Float16 v8h;
typedef __attribute__((ext_vector_type(8)))  float    v8f;

// ---------------------------------------------------------------------------
// Generic WMMA GEMM:  C[M,N] = A[M,K] (f16, row-major, lda) * W[N,K]^T (f16, ldb)
// f32 accumulation via V_WMMA_F32_16X16X32_F16.
// Wave computes a 32x32 tile: 2 A-fragments x 2 B-fragments -> 4 WMMAs per
// K-step on 8 b128 loads (64 B loaded per WMMA vs 96 B for a 16x32 tile).
// 8 waves/block arranged 2(M) x 4(N) -> 64x128 block tile.
// Fragment layout per CDNA5 ISA 7.12.2 (wave32):
//   A (16x32 f16): row M = lane&15; lanes<16 hold K base 0, lanes>=16 base 8;
//                  halves 0..7 -> K = kb+0..7, halves 8..15 -> K = kb+16..23.
//   C/D (16x16 f32, 8 VGPRs): VGPR r -> M = r + (lane>=16 ? 8 : 0), N = lane&15.
// ---------------------------------------------------------------------------
__global__ __launch_bounds__(256) void gemm_f16_wmma(
    const _Float16* __restrict__ A, int lda,
    const _Float16* __restrict__ Bw, int ldb,
    float* __restrict__ C, int ldc,
    int M, int N, int K)
{
    const int lane = threadIdx.x & 31;
    const int wave = threadIdx.x >> 5;
    const int wm   = wave & 1;           // 2 M-tiles of 32
    const int wn   = wave >> 1;          // 4 N-tiles of 32
    const int m0   = blockIdx.y * 64  + wm * 32;
    const int n0   = blockIdx.x * 128 + wn * 32;
    const int hl   = lane >> 4;          // lane half (0/1)
    const int lid  = lane & 15;
    const int kb   = hl * 8;             // K sub-offset per ISA layout

    int ar0 = m0 + lid;      if (ar0 > M - 1) ar0 = M - 1;  // clamp edge rows
    int ar1 = m0 + 16 + lid; if (ar1 > M - 1) ar1 = M - 1;
    int br0 = n0 + lid;      if (br0 > N - 1) br0 = N - 1;
    int br1 = n0 + 16 + lid; if (br1 > N - 1) br1 = N - 1;

    const _Float16* Ap0 = A  + (size_t)ar0 * lda + kb;
    const _Float16* Ap1 = A  + (size_t)ar1 * lda + kb;
    const _Float16* Bp0 = Bw + (size_t)br0 * ldb + kb;
    const _Float16* Bp1 = Bw + (size_t)br1 * ldb + kb;

    v8f acc00 = {}; v8f acc01 = {}; v8f acc10 = {}; v8f acc11 = {};
    for (int k0 = 0; k0 < K; k0 += 32) {
        v8h a0l = *(const v8h*)(Ap0 + k0);
        v8h a0h = *(const v8h*)(Ap0 + k0 + 16);
        v8h a1l = *(const v8h*)(Ap1 + k0);
        v8h a1h = *(const v8h*)(Ap1 + k0 + 16);
        v8h b0l = *(const v8h*)(Bp0 + k0);
        v8h b0h = *(const v8h*)(Bp0 + k0 + 16);
        v8h b1l = *(const v8h*)(Bp1 + k0);
        v8h b1h = *(const v8h*)(Bp1 + k0 + 16);
        v16h af0, af1, bf0, bf1;
#pragma unroll
        for (int i = 0; i < 8; ++i) {
            af0[i] = a0l[i]; af0[i + 8] = a0h[i];
            af1[i] = a1l[i]; af1[i + 8] = a1h[i];
            bf0[i] = b0l[i]; bf0[i + 8] = b0h[i];
            bf1[i] = b1l[i]; bf1[i + 8] = b1h[i];
        }
        acc00 = __builtin_amdgcn_wmma_f32_16x16x32_f16(false, af0, false, bf0,
                                                       (short)0, acc00, false, false);
        acc01 = __builtin_amdgcn_wmma_f32_16x16x32_f16(false, af0, false, bf1,
                                                       (short)0, acc01, false, false);
        acc10 = __builtin_amdgcn_wmma_f32_16x16x32_f16(false, af1, false, bf0,
                                                       (short)0, acc10, false, false);
        acc11 = __builtin_amdgcn_wmma_f32_16x16x32_f16(false, af1, false, bf1,
                                                       (short)0, acc11, false, false);
    }
#pragma unroll
    for (int r = 0; r < 8; ++r) {
        int mr0 = m0 + r + hl * 8;
        int mr1 = mr0 + 16;
        int nc0 = n0 + lid;
        int nc1 = n0 + 16 + lid;
        if (mr0 < M) {
            if (nc0 < N) C[(size_t)mr0 * ldc + nc0] = acc00[r];
            if (nc1 < N) C[(size_t)mr0 * ldc + nc1] = acc01[r];
        }
        if (mr1 < M) {
            if (nc0 < N) C[(size_t)mr1 * ldc + nc0] = acc10[r];
            if (nc1 < N) C[(size_t)mr1 * ldc + nc1] = acc11[r];
        }
    }
}

// ------------------------- elementwise / helper kernels ---------------------

__global__ void f32_to_f16_kernel(const float* __restrict__ s,
                                  _Float16* __restrict__ d, int n)
{
    int i = blockIdx.x * blockDim.x + threadIdx.x;
    if (i < n) d[i] = (_Float16)s[i];
}

// dt_proj_w (DI,24) -> f16 (DI,32), zero-padded K (WMMA needs K multiple of 32)
__global__ void dtw_pad_kernel(const float* __restrict__ s, _Float16* __restrict__ d)
{
    int i = blockIdx.x * blockDim.x + threadIdx.x;
    if (i >= DI_ * 32) return;
    int r = i >> 5, c = i & 31;
    d[i] = (c < DTR_) ? (_Float16)s[r * DTR_ + c] : (_Float16)0.f;
}

// im2col for 16x16/stride-16 patch conv: row=(b,patch), col = c*256 + ky*16 + kx
__global__ void im2col_kernel(const float* __restrict__ x, _Float16* __restrict__ col)
{
    int idx = blockIdx.x * blockDim.x + threadIdx.x;
    if (idx >= B_ * NPATCH * 768) return;
    int k = idx % 768;
    int t = idx / 768;
    int b = t / NPATCH, p = t % NPATCH;
    int c = k / 256, rr = k % 256, ky = rr / 16, kx = rr % 16;
    int py = p / 14, px = p % 14;
    float v = x[(((size_t)b * 3 + c) * 224 + (py * 16 + ky)) * 224 + (px * 16 + kx)];
    col[idx] = (_Float16)v;
}

// hid = [cls | patches + patch_b] + pos_embed
__global__ void assemble_kernel(const float* __restrict__ pout,
                                const float* __restrict__ patch_b,
                                const float* __restrict__ cls,
                                const float* __restrict__ pos,
                                float* __restrict__ hid)
{
    int idx = blockIdx.x * blockDim.x + threadIdx.x;
    if (idx >= NTOK * D_) return;
    int d = idx % D_;
    int t = idx / D_;
    int b = t / L_, l = t % L_;
    float v = pos[l * D_ + d];
    if (l == 0) v += cls[d];
    else        v += pout[((size_t)b * NPATCH + (l - 1)) * D_ + d] + patch_b[d];
    hid[idx] = v;
}

// res += hid; u = LayerNorm(res)*w + b -> f16; optionally pack cls rows (token 0)
__global__ __launch_bounds__(128) void add_ln_kernel(
    float* __restrict__ res, const float* __restrict__ hid,
    const float* __restrict__ w, const float* __restrict__ bb,
    _Float16* __restrict__ out16, _Float16* __restrict__ cls16)
{
    __shared__ float sbuf[128];
    const int tok = blockIdx.x;
    const int t   = threadIdx.x;
    float* r = res + (size_t)tok * D_;
    const float* hh = hid + (size_t)tok * D_;
    float v[3]; float s = 0.f;
#pragma unroll
    for (int i = 0; i < 3; ++i) {
        int d = t + i * 128;
        float u = r[d] + hh[d];
        r[d] = u; v[i] = u; s += u;
    }
    sbuf[t] = s; __syncthreads();
    for (int st = 64; st > 0; st >>= 1) { if (t < st) sbuf[t] += sbuf[t + st]; __syncthreads(); }
    float mu = sbuf[0] * (1.f / D_);
    __syncthreads();
    float s2 = 0.f;
#pragma unroll
    for (int i = 0; i < 3; ++i) { float dd = v[i] - mu; s2 += dd * dd; }
    sbuf[t] = s2; __syncthreads();
    for (int st = 64; st > 0; st >>= 1) { if (t < st) sbuf[t] += sbuf[t + st]; __syncthreads(); }
    float rstd = rsqrtf(sbuf[0] * (1.f / D_) + 1e-5f);
#pragma unroll
    for (int i = 0; i < 3; ++i) {
        int d = t + i * 128;
        float o = (v[i] - mu) * rstd * w[d] + bb[d];
        out16[(size_t)tok * D_ + d] = (_Float16)o;
        if (cls16 && (tok % L_) == 0)
            cls16[(size_t)(tok / L_) * D_ + d] = (_Float16)o;
    }
}

// depthwise causal conv (k=4, left pad 3) over L, + bias, SiLU
__global__ void conv_silu_kernel(const float* __restrict__ xz,
                                 const float* __restrict__ wc,
                                 const float* __restrict__ bc,
                                 float* __restrict__ xc32,
                                 _Float16* __restrict__ xc16)
{
    int idx = blockIdx.x * blockDim.x + threadIdx.x;
    if (idx >= NTOK * DI_) return;
    int d = idx % DI_;
    int t = idx / DI_;          // b*L + l
    int l = t % L_;
    float acc = bc[d];
#pragma unroll
    for (int j = 0; j < 4; ++j) {
        int ls = l + j - 3;
        if (ls >= 0)
            acc += xz[(size_t)(t - l + ls) * (2 * DI_) + d] * wc[d * 4 + j];
    }
    float o = acc / (1.f + expf(-acc));    // SiLU: x*sigmoid(x)
    xc32[idx] = o;
    xc16[idx] = (_Float16)o;
}

// pack dbl[:, :24] into zero-padded f16 (NTOK, 32) for the dt GEMM
__global__ void pack_dt_kernel(const float* __restrict__ dbl, _Float16* __restrict__ dtin)
{
    int idx = blockIdx.x * blockDim.x + threadIdx.x;
    if (idx >= NTOK * 32) return;
    int row = idx >> 5, c = idx & 31;
    dtin[idx] = (c < DTR_) ? (_Float16)dbl[(size_t)row * 56 + c] : (_Float16)0.f;
}

// selective scan: one thread per (batch, channel); DS=16 state in registers.
// Fuses softplus(dt)+bias, recurrence, C-readout, D skip, and SiLU(z) gate.
__global__ void scan_kernel(const float* __restrict__ dtraw,
                            const float* __restrict__ bdt,
                            const float* __restrict__ xc,
                            const float* __restrict__ dbl,
                            const float* __restrict__ xz,
                            const float* __restrict__ Alog,
                            const float* __restrict__ Dp,
                            _Float16* __restrict__ y16)
{
    int idx = blockIdx.x * blockDim.x + threadIdx.x;
    if (idx >= B_ * DI_) return;
    int b = idx / DI_, d = idx % DI_;
    float a[DS_], h[DS_];
#pragma unroll
    for (int n = 0; n < DS_; ++n) { a[n] = -expf(Alog[d * DS_ + n]); h[n] = 0.f; }
    const float bias = bdt[d];
    const float dval = Dp[d];
    for (int l = 0; l < L_; ++l) {
        size_t row = (size_t)b * L_ + l;
        float dt = dtraw[row * DI_ + d] + bias;
        dt = (dt > 20.f) ? dt : log1pf(expf(dt));       // softplus
        float xv  = xc[row * DI_ + d];
        float dtx = dt * xv;
        const float* bcv = dbl + row * 56 + DTR_;       // [B(16) | C(16)]
        float y = 0.f;
#pragma unroll
        for (int n = 0; n < DS_; ++n) {
            float da = expf(dt * a[n]);
            h[n] = da * h[n] + dtx * bcv[n];
            y += h[n] * bcv[DS_ + n];
        }
        y += xv * dval;
        float z = xz[row * (2 * DI_) + DI_ + d];
        y *= z / (1.f + expf(-z));                      // * SiLU(z)
        y16[row * DI_ + d] = (_Float16)y;
    }
}

__global__ void head_bias_kernel(float* __restrict__ out, const float* __restrict__ hb)
{
    int i = blockIdx.x * blockDim.x + threadIdx.x;
    if (i < B_ * NCLS) out[i] += hb[i % NCLS];
}

// ---------------------------------------------------------------------------

extern "C" void kernel_launch(void* const* d_in, const int* in_sizes, int n_in,
                              void* d_out, int out_size, void* d_ws, size_t ws_size,
                              hipStream_t stream)
{
    (void)in_sizes; (void)n_in; (void)out_size; (void)ws_size;
    const float* x         = (const float*)d_in[0];
    const float* patch_w   = (const float*)d_in[1];
    const float* patch_b   = (const float*)d_in[2];
    const float* cls_tok   = (const float*)d_in[3];
    const float* pos_emb   = (const float*)d_in[4];
    const float* in_proj_w = (const float*)d_in[5];
    const float* conv_w    = (const float*)d_in[6];
    const float* conv_b    = (const float*)d_in[7];
    const float* x_proj_w  = (const float*)d_in[8];
    const float* dt_proj_w = (const float*)d_in[9];
    const float* dt_proj_b = (const float*)d_in[10];
    const float* A_log     = (const float*)d_in[11];
    const float* D_ssm     = (const float*)d_in[12];
    const float* out_proj_w= (const float*)d_in[13];
    const float* norm_w    = (const float*)d_in[14];
    const float* norm_b    = (const float*)d_in[15];
    const float* normf_w   = (const float*)d_in[16];
    const float* normf_b   = (const float*)d_in[17];
    const float* head_w    = (const float*)d_in[18];
    const float* head_b    = (const float*)d_in[19];
    float* out = (float*)d_out;

    uint8_t* base = (uint8_t*)d_ws;
    size_t off = 0;
    auto alloc = [&](size_t bytes) -> void* {
        off = (off + 255) & ~(size_t)255;
        void* p = base + off;
        off += bytes;
        return p;
    };

    float*    res    = (float*)   alloc((size_t)NTOK * D_ * 4);
    float*    hid    = (float*)   alloc((size_t)NTOK * D_ * 4);
    _Float16* u16    = (_Float16*)alloc((size_t)NTOK * D_ * 2);
    float*    xz     = (float*)   alloc((size_t)NTOK * 2 * DI_ * 4);
    float*    xc32   = (float*)   alloc((size_t)NTOK * DI_ * 4);
    _Float16* xc16   = (_Float16*)alloc((size_t)NTOK * DI_ * 2);
    float*    dbl    = (float*)   alloc((size_t)NTOK * 56 * 4);
    _Float16* dtin16 = (_Float16*)alloc((size_t)NTOK * 32 * 2);
    float*    dtraw  = (float*)   alloc((size_t)NTOK * DI_ * 4);
    _Float16* y16    = (_Float16*)alloc((size_t)NTOK * DI_ * 2);
    _Float16* col16  = (_Float16*)alloc((size_t)B_ * NPATCH * 768 * 2);
    float*    pout   = (float*)   alloc((size_t)B_ * NPATCH * D_ * 4);
    _Float16* cls16  = (_Float16*)alloc((size_t)B_ * D_ * 2);
    _Float16* wA16   = (_Float16*)alloc((size_t)2 * DI_ * D_ * 2);
    _Float16* wX16   = (_Float16*)alloc((size_t)56 * DI_ * 2);
    _Float16* wDT16  = (_Float16*)alloc((size_t)DI_ * 32 * 2);
    _Float16* wO16   = (_Float16*)alloc((size_t)D_ * DI_ * 2);
    _Float16* pw16   = (_Float16*)alloc((size_t)D_ * 768 * 2);
    _Float16* hw16   = (_Float16*)alloc((size_t)NCLS * D_ * 2);

    auto g1 = [](int n) { return dim3((unsigned)((n + 255) / 256)); };
    auto gemm_grid = [](int M, int N) { return dim3((unsigned)((N + 127) / 128),
                                                    (unsigned)((M + 63) / 64)); };

    // static weight conversions
    f32_to_f16_kernel<<<g1(D_ * 768), 256, 0, stream>>>(patch_w, pw16, D_ * 768);
    f32_to_f16_kernel<<<g1(NCLS * D_), 256, 0, stream>>>(head_w, hw16, NCLS * D_);

    // patch embedding (im2col + WMMA GEMM), token assembly
    im2col_kernel<<<g1(B_ * NPATCH * 768), 256, 0, stream>>>(x, col16);
    gemm_f16_wmma<<<gemm_grid(B_ * NPATCH, D_), 256, 0, stream>>>(
        col16, 768, pw16, 768, pout, D_, B_ * NPATCH, D_, 768);
    assemble_kernel<<<g1(NTOK * D_), 256, 0, stream>>>(pout, patch_b, cls_tok, pos_emb, hid);
    hipMemsetAsync(res, 0, (size_t)NTOK * D_ * 4, stream);

    for (int layer = 0; layer < DEPTH_; ++layer) {
        const float* Win  = in_proj_w  + (size_t)layer * 2 * DI_ * D_;
        const float* Wc   = conv_w     + (size_t)layer * DI_ * 4;
        const float* bc   = conv_b     + (size_t)layer * DI_;
        const float* Wx   = x_proj_w   + (size_t)layer * 56 * DI_;
        const float* Wdt  = dt_proj_w  + (size_t)layer * DI_ * DTR_;
        const float* bdt  = dt_proj_b  + (size_t)layer * DI_;
        const float* Al   = A_log      + (size_t)layer * DI_ * DS_;
        const float* Dp   = D_ssm      + (size_t)layer * DI_;
        const float* Wout = out_proj_w + (size_t)layer * D_ * DI_;
        const float* nw   = norm_w     + (size_t)layer * D_;
        const float* nb   = norm_b     + (size_t)layer * D_;

        // per-layer weight -> f16 staging (reused buffers keep ws small)
        f32_to_f16_kernel<<<g1(2 * DI_ * D_), 256, 0, stream>>>(Win, wA16, 2 * DI_ * D_);
        f32_to_f16_kernel<<<g1(56 * DI_), 256, 0, stream>>>(Wx, wX16, 56 * DI_);
        dtw_pad_kernel<<<g1(DI_ * 32), 256, 0, stream>>>(Wdt, wDT16);
        f32_to_f16_kernel<<<g1(D_ * DI_), 256, 0, stream>>>(Wout, wO16, D_ * DI_);

        // res += hid; u = LN(res)
        add_ln_kernel<<<NTOK, 128, 0, stream>>>(res, hid, nw, nb, u16, (_Float16*)nullptr);

        // xz = u @ Win^T   (M=3152, N=1536, K=384)
        gemm_f16_wmma<<<gemm_grid(NTOK, 2 * DI_), 256, 0, stream>>>(
            u16, D_, wA16, D_, xz, 2 * DI_, NTOK, 2 * DI_, D_);

        // depthwise causal conv + SiLU
        conv_silu_kernel<<<g1(NTOK * DI_), 256, 0, stream>>>(xz, Wc, bc, xc32, xc16);

        // dbl = xc @ Wx^T  (N=56, edge-guarded)
        gemm_f16_wmma<<<gemm_grid(NTOK, 56), 256, 0, stream>>>(
            xc16, DI_, wX16, DI_, dbl, 56, NTOK, 56, DI_);

        // dtraw = dbl[:, :24] @ Wdt^T (K padded to 32)
        pack_dt_kernel<<<g1(NTOK * 32), 256, 0, stream>>>(dbl, dtin16);
        gemm_f16_wmma<<<gemm_grid(NTOK, DI_), 256, 0, stream>>>(
            dtin16, 32, wDT16, 32, dtraw, DI_, NTOK, DI_, 32);

        // fused selective scan + gating -> y16
        scan_kernel<<<g1(B_ * DI_), 256, 0, stream>>>(dtraw, bdt, xc32, dbl, xz, Al, Dp, y16);

        // hid = y @ Wout^T (M=3152, N=384, K=768)
        gemm_f16_wmma<<<gemm_grid(NTOK, D_), 256, 0, stream>>>(
            y16, DI_, wO16, DI_, hid, D_, NTOK, D_, DI_);
    }

    // final LN (also packs cls rows) + head GEMM + bias
    add_ln_kernel<<<NTOK, 128, 0, stream>>>(res, hid, normf_w, normf_b, u16, cls16);
    gemm_f16_wmma<<<gemm_grid(B_, NCLS), 256, 0, stream>>>(
        cls16, D_, hw16, D_, out, NCLS, B_, NCLS, D_);
    head_bias_kernel<<<g1(B_ * NCLS), 256, 0, stream>>>(out, head_b);
}
